// GridGNNWithAngles_44504451121306
// MI455X (gfx1250) — matrix operand order repaired and checked by reading, not compiled, not run on hardware
//
#include <hip/hip_runtime.h>
#include <hip/hip_bf16.h>

// ---------------------------------------------------------------------------
// GridGNNWithAngles on MI455X (gfx1250, wave32, WMMA bf16 path)
// ---------------------------------------------------------------------------
#define N_NODES 50000
#define N_EDGES 800000
#define CH 64
#define WAVES_PER_BLOCK 8
#define TILES (N_EDGES / 16)          // 50000, exact
#define EDGE_BLOCKS (TILES / WAVES_PER_BLOCK)  // 6250, exact

typedef __attribute__((ext_vector_type(16))) __bf16 v16bf;
typedef __attribute__((ext_vector_type(8)))  float  v8f;

__device__ __forceinline__ unsigned short f2bf(float f) {
    unsigned u = __float_as_uint(f);
    unsigned r = u + 0x7FFFu + ((u >> 16) & 1u);   // round-to-nearest-even
    return (unsigned short)(r >> 16);
}

__device__ __forceinline__ float fast_tanh(float v) {
#if __has_builtin(__builtin_amdgcn_tanhf)
    return __builtin_amdgcn_tanhf(v);     // single V_TANH_F32 trans op
#else
    return tanhf(v);
#endif
}

// --- prep: sin/cos per edge (hardware v_sin/v_cos) --------------------------
__global__ void sincos_kernel(const float* __restrict__ attr,
                              float2* __restrict__ sc, int n) {
    int i = blockIdx.x * blockDim.x + threadIdx.x;
    if (i >= n) return;
    float rad = attr[i] * 0.017453292519943295f;
    sc[i] = make_float2(__sinf(rad), __cosf(rad));
}

// --- prep: f32 -> bf16 node features ---------------------------------------
__global__ void cvt_bf16_kernel(const float* __restrict__ in,
                                unsigned short* __restrict__ out, int n) {
    int i = blockIdx.x * blockDim.x + threadIdx.x;
    if (i >= n) return;
    out[i] = f2bf(in[i]);
}

// --- prep: zero an f32 buffer ----------------------------------------------
__global__ void zero_kernel(float* __restrict__ p, int n) {
    int i = blockIdx.x * blockDim.x + threadIdx.x;
    if (i < n) p[i] = 0.0f;
}

// --- prep: pack W[0:64,0:64] into B-fragment lane layout --------------------
// frag = kt*4 + nt ; element addr = frag*512 + lane*16 + h  (bf16 units)
// lane: col = lane&15 ; K = kt*32 + (h&7) + (lane>=16 ? 8 : 0) + (h>=8 ? 16 : 0)
__global__ void pack_w_kernel(const float* __restrict__ Wf,
                              unsigned short* __restrict__ pW) {
    int i = blockIdx.x * blockDim.x + threadIdx.x;
    if (i >= 4096) return;
    int h    = i & 15;
    int lane = (i >> 4) & 31;
    int nt   = (i >> 9) & 3;
    int kt   = (i >> 11) & 1;
    int K = kt * 32 + (h & 7) + ((lane >= 16) ? 8 : 0) + ((h >= 8) ? 16 : 0);
    int N = nt * 16 + (lane & 15);
    pW[i] = f2bf(Wf[K * CH + N]);
}

// --- main edge kernel: gather -> WMMA GEMM -> (tanh) -> atomic scatter ------
// Indices are int64 in memory but < 2^31, so we read only the low DWORD and
// keep all address math 32-bit (GVS addressing: SGPR base + VGPR offset).
template <int APPLY_TANH>
__global__ __launch_bounds__(32 * WAVES_PER_BLOCK)
void edge_layer_kernel(const unsigned short* __restrict__ xbf,   // [N,64] bf16
                       const int* __restrict__ src32,            // [E] stride-2 (int64 lo)
                       const int* __restrict__ dst32,            // [E] stride-2 (int64 lo)
                       const float2* __restrict__ sincos,        // [E]
                       const unsigned short* __restrict__ pW,    // packed B frags
                       const float* __restrict__ Wf,             // [66,64] f32
                       const float* __restrict__ bvec,           // [64]
                       float* __restrict__ agg) {                // [N,64] f32 (atomic)
    __shared__ unsigned short ldsA[WAVES_PER_BLOCK][16 * CH];   // 2KB / wave

    const int lane = threadIdx.x & 31;
    const int wave = threadIdx.x >> 5;
    const int tile = blockIdx.x * WAVES_PER_BLOCK + wave;
    const int e0   = tile * 16;
    const int row  = lane & 15;
    const int half = lane >> 4;

    // prefetch next block's indices (global_prefetch_b8)
    __builtin_prefetch(src32 + 2 * (e0 + 16 * WAVES_PER_BLOCK), 0, 0);

    // ---- gather: 16 rows x 128B into LDS, 4 x b128 per lane ----
    {
        int sIdx = src32[2 * (e0 + row)];
        const uint4* srow = (const uint4*)(xbf + (size_t)((unsigned)sIdx) * CH);
        uint4* lrow = (uint4*)(&ldsA[wave][row * CH]);
        #pragma unroll
        for (int i = 0; i < 4; ++i) {
            int part = half * 4 + i;
            lrow[part] = srow[part];
        }
    }
    __syncthreads();

    // ---- build A fragments (16x32 bf16, two K-tiles) per ISA layout ----
    // lanes 0-15 : halfs 0-7 -> K kb+0..7,  halfs 8-15 -> K kb+16..23
    // lanes 16-31: halfs 0-7 -> K kb+8..15, halfs 8-15 -> K kb+24..31
    v16bf a0, a1;
    {
        const unsigned short* base = &ldsA[wave][row * CH];
        union { v16bf v; uint4 q[2]; } ua;
        ua.q[0] = *(const uint4*)(base + 0  + half * 8);
        ua.q[1] = *(const uint4*)(base + 16 + half * 8);
        a0 = ua.v;
        ua.q[0] = *(const uint4*)(base + 32 + half * 8);
        ua.q[1] = *(const uint4*)(base + 48 + half * 8);
        a1 = ua.v;
    }

    // ---- B fragments (32x16 bf16), pre-packed: 2 b128 loads each ----
    v16bf Bf[2][4];
    #pragma unroll
    for (int kt = 0; kt < 2; ++kt)
        #pragma unroll
        for (int nt = 0; nt < 4; ++nt) {
            const uint4* p = (const uint4*)(pW + ((kt * 4 + nt) * 512 + lane * 16));
            union { v16bf v; uint4 q[2]; } ub;
            ub.q[0] = p[0];
            ub.q[1] = p[1];
            Bf[kt][nt] = ub.v;
        }

    // ---- per-row edge data for C-init and scatter (M = r + half*8) ----
    float sinM[8], cosM[8];
    int   dstOff[8];                           // 32-bit element offsets into agg
    #pragma unroll
    for (int r = 0; r < 8; ++r) {
        int M = r + half * 8;
        float2 s2 = sincos[e0 + M];
        sinM[r] = s2.x;
        cosM[r] = s2.y;
        dstOff[r] = dst32[2 * (e0 + M)] * CH;  // < 3.2M, fits int
    }

    const int ncol = lane & 15;
    #pragma unroll
    for (int nt = 0; nt < 4; ++nt) {
        int N = nt * 16 + ncol;
        float bN  = bvec[N];
        float w64 = Wf[64 * CH + N];    // sin weight row
        float w65 = Wf[65 * CH + N];    // cos weight row
        v8f acc;
        #pragma unroll
        for (int r = 0; r < 8; ++r)
            acc[r] = bN + sinM[r] * w64 + cosM[r] * w65;

        acc = __builtin_amdgcn_wmma_f32_16x16x32_bf16(
                  false, a0, false, Bf[0][nt], (short)0, acc, false, false);
        acc = __builtin_amdgcn_wmma_f32_16x16x32_bf16(
                  false, a1, false, Bf[1][nt], (short)0, acc, false, false);

        #pragma unroll
        for (int r = 0; r < 8; ++r) {
            float v = acc[r];
            if (APPLY_TANH) v = fast_tanh(v);
            atomicAdd(agg + (dstOff[r] + N), v);
        }
    }
}

// --- epilogue 1: h = bf16(relu(agg + bias1)) -------------------------------
__global__ void finish1_kernel(const float* __restrict__ agg,
                               const float* __restrict__ bias,
                               unsigned short* __restrict__ hbf, int n) {
    int i = blockIdx.x * blockDim.x + threadIdx.x;
    if (i >= n) return;
    float v = agg[i] + bias[i & (CH - 1)];
    v = fmaxf(v, 0.0f);
    hbf[i] = f2bf(v);
}

// --- epilogue 2: out += bias2 ----------------------------------------------
__global__ void finish2_kernel(float* __restrict__ out,
                               const float* __restrict__ bias, int n) {
    int i = blockIdx.x * blockDim.x + threadIdx.x;
    if (i >= n) return;
    out[i] += bias[i & (CH - 1)];
}

extern "C" void kernel_launch(void* const* d_in, const int* in_sizes, int n_in,
                              void* d_out, int out_size, void* d_ws, size_t ws_size,
                              hipStream_t stream) {
    (void)in_sizes; (void)n_in; (void)out_size; (void)ws_size;

    const float* x     = (const float*)d_in[0];
    const int*   eidx  = (const int*)d_in[1];      // [2,E] int64, read lo DWORDs
    const float* attr  = (const float*)d_in[2];
    const float* W1    = (const float*)d_in[3];    // [66,64]
    const float* b1    = (const float*)d_in[4];
    const float* bias1 = (const float*)d_in[5];
    const float* W2    = (const float*)d_in[6];
    const float* b2    = (const float*)d_in[7];
    const float* bias2 = (const float*)d_in[8];
    float*       out   = (float*)d_out;

    const int* src32 = eidx;                       // int64 lanes, stride 2
    const int* dst32 = eidx + 2 * (size_t)N_EDGES; // second row of edge_index

    // workspace layout (256B aligned)
    char* ws = (char*)d_ws;
    size_t off = 0;
    auto take = [&](size_t bytes) { size_t o = off; off += (bytes + 255) & ~(size_t)255; return o; };
    float2*         sc   = (float2*)        (ws + take((size_t)N_EDGES * 8));
    unsigned short* xbf  = (unsigned short*)(ws + take((size_t)N_NODES * CH * 2));
    unsigned short* hbf  = (unsigned short*)(ws + take((size_t)N_NODES * CH * 2));
    float*          agg1 = (float*)         (ws + take((size_t)N_NODES * CH * 4));
    unsigned short* pW1  = (unsigned short*)(ws + take(4096 * 2));
    unsigned short* pW2  = (unsigned short*)(ws + take(4096 * 2));

    const int TPB = 256;
    const int nFeat = N_NODES * CH;

    // prep
    sincos_kernel  <<<(N_EDGES + TPB - 1) / TPB, TPB, 0, stream>>>(attr, sc, N_EDGES);
    cvt_bf16_kernel<<<(nFeat + TPB - 1) / TPB, TPB, 0, stream>>>(x, xbf, nFeat);
    pack_w_kernel  <<<16, TPB, 0, stream>>>(W1, pW1);
    pack_w_kernel  <<<16, TPB, 0, stream>>>(W2, pW2);
    zero_kernel    <<<(nFeat + TPB - 1) / TPB, TPB, 0, stream>>>(agg1, nFeat);
    zero_kernel    <<<(nFeat + TPB - 1) / TPB, TPB, 0, stream>>>(out, nFeat);

    // layer 1: gather/GEMM/tanh/scatter
    edge_layer_kernel<1><<<EDGE_BLOCKS, 32 * WAVES_PER_BLOCK, 0, stream>>>(
        xbf, src32, dst32, sc, pW1, W1, b1, agg1);
    finish1_kernel<<<(nFeat + TPB - 1) / TPB, TPB, 0, stream>>>(agg1, bias1, hbf, nFeat);

    // layer 2: gather/GEMM/scatter (no tanh)
    edge_layer_kernel<0><<<EDGE_BLOCKS, 32 * WAVES_PER_BLOCK, 0, stream>>>(
        hbf, src32, dst32, sc, pW2, W2, b2, out);
    finish2_kernel<<<(nFeat + TPB - 1) / TPB, TPB, 0, stream>>>(out, bias2, nFeat);
}
